// Block_1632087573278
// MI455X (gfx1250) — matrix operand order repaired and checked by reading, not compiled, not run on hardware
//
#include <hip/hip_runtime.h>
#include <hip/hip_bf16.h>

// ---------------------------------------------------------------------------
// Types
// ---------------------------------------------------------------------------
typedef unsigned short u16;
typedef unsigned int   u32;
typedef __attribute__((ext_vector_type(4)))  u32    u32x4;
typedef __attribute__((ext_vector_type(16))) __bf16 bf16x16;
typedef __attribute__((ext_vector_type(8)))  float  f32x8;

union FragB {            // one WMMA bf16 operand: 16 bf16 = 8 VGPRs
    u32x4   u[2];
    bf16x16 v;
};
union U16x8 { u32x4 q; u16 s[8]; };

constexpr int Dm   = 1024;
constexpr int Sq   = 2048;
constexpr int NBat = 2;
constexpr int NH   = 16;
constexpr int Mtok = NBat * Sq;      // 4096 rows

__device__ __forceinline__ u16 f2bf(float f) {
    u32 x = __builtin_bit_cast(u32, f);
    x += 0x7FFFu + ((x >> 16) & 1u);          // round-to-nearest-even
    return (u16)(x >> 16);
}

__device__ __forceinline__ f32x8 wmma_bf16(const FragB& a, const FragB& b, f32x8 c) {
    // v_wmma_f32_16x16x32_bf16  (neg_a, A, neg_b, B, c_mod, C, reuse_a, reuse_b)
    return __builtin_amdgcn_wmma_f32_16x16x32_bf16(false, a.v, false, b.v,
                                                   (short)0, c, false, false);
}

// ---------------------------------------------------------------------------
// fp32 -> bf16 conversion (weights), grid-stride
// ---------------------------------------------------------------------------
__global__ void cvt_f32_bf16(const float* __restrict__ in, u16* __restrict__ out, int n) {
    int i = blockIdx.x * blockDim.x + threadIdx.x;
    int stride = gridDim.x * blockDim.x;
    for (; i < n; i += stride) out[i] = f2bf(in[i]);
}

// ---------------------------------------------------------------------------
// LayerNorm over D=1024, one row per block (256 threads x 4 elems), bf16 out
// ---------------------------------------------------------------------------
__global__ __launch_bounds__(256) void ln_kernel(const float* __restrict__ x,
                                                 const float* __restrict__ g,
                                                 const float* __restrict__ b,
                                                 u16* __restrict__ out) {
    const int row = blockIdx.x;
    const int t   = threadIdx.x;
    const float4 v = ((const float4*)(x + (size_t)row * Dm))[t];

    __shared__ float rs[256], rss[256];
    rs[t]  = v.x + v.y + v.z + v.w;
    rss[t] = v.x * v.x + v.y * v.y + v.z * v.z + v.w * v.w;
    __syncthreads();
#pragma unroll
    for (int o = 128; o > 0; o >>= 1) {
        if (t < o) { rs[t] += rs[t + o]; rss[t] += rss[t + o]; }
        __syncthreads();
    }
    const float mu  = rs[0] * (1.0f / Dm);
    const float var = rss[0] * (1.0f / Dm) - mu * mu;
    const float inv = rsqrtf(var + 1e-5f);

    const int i0 = t * 4;
    float e[4] = {v.x, v.y, v.z, v.w};
#pragma unroll
    for (int j = 0; j < 4; ++j)
        out[(size_t)row * Dm + i0 + j] = f2bf((e[j] - mu) * inv * g[i0 + j] + b[i0 + j]);
}

// ---------------------------------------------------------------------------
// bf16 WMMA GEMM:  C[M,N] = A[M,K] * W[N,K]^T + bias  (+gelu) (+resid)
// Block tile 128x256, 8 waves (2x4), wave tile 64x64 (16 WMMA / K-step),
// BK = 32 (one WMMA step), double-buffered LDS, software-pipelined loads.
// ---------------------------------------------------------------------------
__global__ __launch_bounds__(256) void gemm_bf16_kernel(
        const u16* __restrict__ A, const u16* __restrict__ W,
        const float* __restrict__ bias, const float* __restrict__ resid,
        float* __restrict__ outF, u16* __restrict__ outB,
        int M, int N, int K, int dogelu) {
    __shared__ __align__(16) u16 As[2][128][32];   // 16 KB
    __shared__ __align__(16) u16 Bs[2][256][32];   // 32 KB

    const int tid  = threadIdx.x;
    const int lane = tid & 31;
    const int w    = tid >> 5;
    const int wrow = w >> 2;          // 0..1 -> 64 rows each
    const int wcol = w & 3;           // 0..3 -> 64 cols each
    const int m0   = blockIdx.y * 128;
    const int n0   = blockIdx.x * 256;

    f32x8 acc[4][4];
#pragma unroll
    for (int mi = 0; mi < 4; ++mi)
#pragma unroll
        for (int ni = 0; ni < 4; ++ni) acc[mi][ni] = (f32x8){0,0,0,0,0,0,0,0};

    const int koff = (lane >> 4) * 8;     // A-layout K split per lane half
    const int kr   = (lane >> 4) * 16;    // unused; B fragments also load with koff

    // staging-chunk geometry: 8-elem (16B) chunks, 4 chunks per 32-wide row
    int arow[2], akp[2], brow[4], bkp[4];
#pragma unroll
    for (int i = 0; i < 2; ++i) {           // A: 128x32 = 512 chunks
        const int c = tid + i * 256; arow[i] = c >> 2; akp[i] = (c & 3) * 8;
    }
#pragma unroll
    for (int i = 0; i < 4; ++i) {           // B: 256x32 = 1024 chunks
        const int c = tid + i * 256; brow[i] = c >> 2; bkp[i] = (c & 3) * 8;
    }

    u32x4 ra[2], rb[4];
    // ---- prologue: load tile 0 and stage into buffer 0 --------------------
#pragma unroll
    for (int i = 0; i < 2; ++i)
        ra[i] = *(const u32x4*)(A + (size_t)(m0 + arow[i]) * K + akp[i]);
#pragma unroll
    for (int i = 0; i < 4; ++i)
        rb[i] = *(const u32x4*)(W + (size_t)(n0 + brow[i]) * K + bkp[i]);
#pragma unroll
    for (int i = 0; i < 2; ++i) *(u32x4*)&As[0][arow[i]][akp[i]] = ra[i];
#pragma unroll
    for (int i = 0; i < 4; ++i) *(u32x4*)&Bs[0][brow[i]][bkp[i]] = rb[i];
    __syncthreads();

    const int T = K >> 5;
    for (int t = 0; t < T; ++t) {
        const int cur = t & 1;
        // ---- issue next tile's global loads (latency hidden by compute) ---
        if (t + 1 < T) {
            const int kk = (t + 1) << 5;
#pragma unroll
            for (int i = 0; i < 2; ++i) {
                const u16* ga = A + (size_t)(m0 + arow[i]) * K + kk + akp[i];
                ra[i] = *(const u32x4*)ga;
                __builtin_prefetch(ga + 32, 0, 1);   // speculative, always on
            }
#pragma unroll
            for (int i = 0; i < 4; ++i) {
                const u16* gb = W + (size_t)(n0 + brow[i]) * K + kk + bkp[i];
                rb[i] = *(const u32x4*)gb;
                __builtin_prefetch(gb + 32, 0, 1);
            }
        }
        // ---- compute current tile: 16 WMMAs from 8 fragment loads ---------
        FragB af[4], bf[4];
#pragma unroll
        for (int mi = 0; mi < 4; ++mi) {
            const int r = wrow * 64 + mi * 16 + (lane & 15);
            af[mi].u[0] = *(const u32x4*)&As[cur][r][koff];
            af[mi].u[1] = *(const u32x4*)&As[cur][r][koff + 16];
        }
#pragma unroll
        for (int ni = 0; ni < 4; ++ni) {
            const int n = wcol * 64 + ni * 16 + (lane & 15);
            bf[ni].u[0] = *(const u32x4*)&Bs[cur][n][koff];
            bf[ni].u[1] = *(const u32x4*)&Bs[cur][n][koff + 16];
        }
#pragma unroll
        for (int mi = 0; mi < 4; ++mi)
#pragma unroll
            for (int ni = 0; ni < 4; ++ni)
                acc[mi][ni] = wmma_bf16(af[mi], bf[ni], acc[mi][ni]);

        // ---- stage next tile into the other buffer ------------------------
        if (t + 1 < T) {
            const int nxt = cur ^ 1;
            __syncthreads();
#pragma unroll
            for (int i = 0; i < 2; ++i) *(u32x4*)&As[nxt][arow[i]][akp[i]] = ra[i];
#pragma unroll
            for (int i = 0; i < 4; ++i) *(u32x4*)&Bs[nxt][brow[i]][bkp[i]] = rb[i];
            __syncthreads();
        }
    }

    // Epilogue: C layout -> row = 8*(lane>=16)+vgpr, col = lane&15
#pragma unroll
    for (int mi = 0; mi < 4; ++mi)
#pragma unroll
        for (int ni = 0; ni < 4; ++ni)
#pragma unroll
            for (int r = 0; r < 8; ++r) {
                const int row = m0 + wrow * 64 + mi * 16 + ((lane >> 4) << 3) + r;
                const int col = n0 + wcol * 64 + ni * 16 + (lane & 15);
                float vv = acc[mi][ni][r] + bias[col];
                if (dogelu) {
                    const float x3 = vv * vv * vv;
                    vv = 0.5f * vv * (1.0f + tanhf(0.7978845608f * (vv + 0.044715f * x3)));
                }
                const size_t idx = (size_t)row * N + col;
                if (resid) vv += resid[idx];
                if (outF) outF[idx] = vv; else outB[idx] = f2bf(vv);
            }
}

// ---------------------------------------------------------------------------
// Flash attention: one block per (b, h, 128-row q tile); 8 waves x 16 q rows.
// QK^T and PV on v_wmma_f32_16x16x32_bf16; online softmax with wave32
// shuffle reductions across the 16-lane C-matrix halves.
// qkv layout: [Mtok, 3*Dm], head h occupies cols h*64..h*64+63 in each third.
// ---------------------------------------------------------------------------
__global__ __launch_bounds__(256) void attn_kernel(const u16* __restrict__ qkv,
                                                   u16* __restrict__ o) {
    __shared__ __align__(16) u16 Ks[64][64];       // [kv][dh]
    __shared__ __align__(16) u16 Vs[64][64];       // transposed: [dh][kv]
    __shared__ __align__(16) u16 Ps[8][16][64];    // per-wave P staging

    const int sb   = Sq / 128;                     // 16 q tiles per (b,h)
    const int qblk = blockIdx.x % sb;
    const int h    = (blockIdx.x / sb) % NH;
    const int bidx = blockIdx.x / (sb * NH);

    const int tid  = threadIdx.x;
    const int lane = tid & 31;
    const int w    = tid >> 5;
    const int qr0  = qblk * 128 + w * 16;

    // --- Q fragments (A operand), 2 chunks over DH=64, kept in registers ---
    FragB qf[2];
    {
        const int qrow = qr0 + (lane & 15);
        const int koff = (lane >> 4) * 8;
        const u16* qb  = qkv + (size_t)(bidx * Sq + qrow) * (3 * Dm) + h * 64;
#pragma unroll
        for (int kc = 0; kc < 2; ++kc) {
            qf[kc].u[0] = *(const u32x4*)(qb + kc * 32 + koff);
            qf[kc].u[1] = *(const u32x4*)(qb + kc * 32 + koff + 16);
        }
    }

    f32x8 otile[4];
#pragma unroll
    for (int dt = 0; dt < 4; ++dt) otile[dt] = (f32x8){0,0,0,0,0,0,0,0};
    float mprev[8], lsum[8];
#pragma unroll
    for (int r = 0; r < 8; ++r) { mprev[r] = -1e30f; lsum[r] = 0.0f; }

    const int nb = 2 * qblk + 2;                   // causal kv-block count
    for (int kb = 0; kb < nb; ++kb) {
        const int kv0 = kb * 64;
        __syncthreads();                            // prior Ks/Vs reads done
        // ---- stage K (natural) and V (transposed) tiles: 64x64 bf16 each --
#pragma unroll
        for (int i = 0; i < 2; ++i) {
            const int chunk = tid + i * 256;        // 512 chunks of 8 elems
            const int row   = chunk >> 3;
            const int cp    = (chunk & 7) * 8;
            const u16* base = qkv + (size_t)(bidx * Sq + kv0 + row) * (3 * Dm) + h * 64;
            *(u32x4*)&Ks[row][cp] = *(const u32x4*)(base + Dm + cp);     // K
            U16x8 vv; vv.q = *(const u32x4*)(base + 2 * Dm + cp);        // V
#pragma unroll
            for (int j = 0; j < 8; ++j) Vs[cp + j][row] = vv.s[j];
        }
        __syncthreads();

        // ---- S = Q K^T * (1/sqrt(64)) -----------------------------------
        float stv[4][8];
#pragma unroll
        for (int nt = 0; nt < 4; ++nt) {
            f32x8 sa = (f32x8){0,0,0,0,0,0,0,0};
            const int col = nt * 16 + (lane & 15);
#pragma unroll
            for (int kc = 0; kc < 2; ++kc) {
                FragB kbf;
                const u16* kp = &Ks[col][kc * 32 + ((lane >> 4) << 3)];
                kbf.u[0] = *(const u32x4*)kp;
                kbf.u[1] = *(const u32x4*)(kp + 16);
                sa = wmma_bf16(qf[kc], kbf, sa);
            }
#pragma unroll
            for (int r = 0; r < 8; ++r) {
                float sv = sa[r] * 0.125f;
                const int kvc = kv0 + nt * 16 + (lane & 15);
                const int qr  = qr0 + ((lane >> 4) << 3) + r;
                stv[nt][r] = (kvc > qr) ? -1e30f : sv;      // causal mask
            }
        }

        // ---- online softmax (row = per-half-wave, reduce over 16 lanes) --
#pragma unroll
        for (int r = 0; r < 8; ++r) {
            float mx = stv[0][r];
#pragma unroll
            for (int nt = 1; nt < 4; ++nt) mx = fmaxf(mx, stv[nt][r]);
#pragma unroll
            for (int off = 1; off < 16; off <<= 1)
                mx = fmaxf(mx, __shfl_xor(mx, off, 32));
            const float nm = fmaxf(mprev[r], mx);
            const float al = __expf(mprev[r] - nm);
            mprev[r] = nm;
            lsum[r] *= al;
#pragma unroll
            for (int dt = 0; dt < 4; ++dt) otile[dt][r] *= al;
        }
        float rsum[8];
#pragma unroll
        for (int r = 0; r < 8; ++r) rsum[r] = 0.0f;
#pragma unroll
        for (int nt = 0; nt < 4; ++nt)
#pragma unroll
            for (int r = 0; r < 8; ++r) {
                const float p = __expf(stv[nt][r] - mprev[r]);
                rsum[r] += p;
                Ps[w][((lane >> 4) << 3) + r][nt * 16 + (lane & 15)] = f2bf(p);
            }
#pragma unroll
        for (int r = 0; r < 8; ++r) {
            float s = rsum[r];
#pragma unroll
            for (int off = 1; off < 16; off <<= 1) s += __shfl_xor(s, off, 32);
            lsum[r] += s;
        }
        __syncthreads();                            // P staged (in-order DS)

        // ---- O += P V  (P as A operand from LDS, V^T as B operand) -------
#pragma unroll
        for (int dt = 0; dt < 4; ++dt) {
            f32x8 oa = otile[dt];
#pragma unroll
            for (int kc = 0; kc < 2; ++kc) {
                FragB pf, vb;
                const u16* pp = &Ps[w][lane & 15][kc * 32 + ((lane >> 4) << 3)];
                pf.u[0] = *(const u32x4*)pp;
                pf.u[1] = *(const u32x4*)(pp + 16);
                const u16* vp = &Vs[dt * 16 + (lane & 15)][kc * 32 + ((lane >> 4) << 3)];
                vb.u[0] = *(const u32x4*)vp;
                vb.u[1] = *(const u32x4*)(vp + 16);
                oa = wmma_bf16(pf, vb, oa);
            }
            otile[dt] = oa;
        }
    }

    // ---- write O / l ------------------------------------------------------
#pragma unroll
    for (int dt = 0; dt < 4; ++dt)
#pragma unroll
        for (int r = 0; r < 8; ++r) {
            const int qr = qr0 + ((lane >> 4) << 3) + r;
            const size_t m = (size_t)bidx * Sq + qr;
            const float val = otile[dt][r] / lsum[r];
            o[m * Dm + h * 64 + dt * 16 + (lane & 15)] = f2bf(val);
        }
}

// ---------------------------------------------------------------------------
// Host-side orchestration
// ---------------------------------------------------------------------------
extern "C" void kernel_launch(void* const* d_in, const int* in_sizes, int n_in,
                              void* d_out, int out_size, void* d_ws, size_t ws_size,
                              hipStream_t stream) {
    const float* x      = (const float*)d_in[0];
    const float* ln1_g  = (const float*)d_in[1];
    const float* ln1_b  = (const float*)d_in[2];
    const float* W_attn = (const float*)d_in[3];
    const float* b_attn = (const float*)d_in[4];
    const float* W_proj = (const float*)d_in[5];
    const float* b_proj = (const float*)d_in[6];
    const float* ln2_g  = (const float*)d_in[7];
    const float* ln2_b  = (const float*)d_in[8];
    const float* W_up   = (const float*)d_in[9];
    const float* b_up   = (const float*)d_in[10];
    const float* W_down = (const float*)d_in[11];
    const float* b_down = (const float*)d_in[12];
    float* out = (float*)d_out;

    char* ws = (char*)d_ws;
    size_t off = 0;
    auto alloc = [&](size_t elems, size_t esz) {
        void* p = ws + off;
        off += (elems * esz + 255) & ~(size_t)255;
        return p;
    };
    u16*   wA   = (u16*)  alloc((size_t)3 * Dm * Dm, 2);   // W_attn bf16
    u16*   wP   = (u16*)  alloc((size_t)Dm * Dm, 2);       // W_proj bf16
    u16*   wU   = (u16*)  alloc((size_t)4 * Dm * Dm, 2);   // W_up bf16
    u16*   wD   = (u16*)  alloc((size_t)4 * Dm * Dm, 2);   // W_down bf16
    u16*   h1   = (u16*)  alloc((size_t)Mtok * Dm, 2);     // ln1(x) bf16
    u16*   qkv  = (u16*)  alloc((size_t)Mtok * 3 * Dm, 2); // qkv bf16
    u16*   obuf = (u16*)  alloc((size_t)Mtok * Dm, 2);     // attn out bf16
    float* x1   = (float*)alloc((size_t)Mtok * Dm, 4);     // x + proj (fp32)
    u16*   h2   = (u16*)  alloc((size_t)Mtok * Dm, 2);     // ln2(x1) bf16
    u16*   ubuf = (u16*)  alloc((size_t)Mtok * 4 * Dm, 2); // gelu(up) bf16
    (void)ws_size; (void)in_sizes; (void)n_in; (void)out_size;

    // 1) weights -> bf16
    cvt_f32_bf16<<<1024, 256, 0, stream>>>(W_attn, wA, 3 * Dm * Dm);
    cvt_f32_bf16<<<1024, 256, 0, stream>>>(W_proj, wP, Dm * Dm);
    cvt_f32_bf16<<<1024, 256, 0, stream>>>(W_up,   wU, 4 * Dm * Dm);
    cvt_f32_bf16<<<1024, 256, 0, stream>>>(W_down, wD, 4 * Dm * Dm);

    // 2) h1 = LN1(x)
    ln_kernel<<<Mtok, 256, 0, stream>>>(x, ln1_g, ln1_b, h1);

    // 3) qkv = h1 @ W_attn^T + b_attn          (4096 x 3072 x 1024)
    gemm_bf16_kernel<<<dim3(3 * Dm / 256, Mtok / 128), 256, 0, stream>>>(
        h1, wA, b_attn, nullptr, nullptr, qkv, Mtok, 3 * Dm, Dm, 0);

    // 4) flash attention
    attn_kernel<<<NBat * NH * (Sq / 128), 256, 0, stream>>>(qkv, obuf);

    // 5) x1 = x + o @ W_proj^T + b_proj        (4096 x 1024 x 1024)
    gemm_bf16_kernel<<<dim3(Dm / 256, Mtok / 128), 256, 0, stream>>>(
        obuf, wP, b_proj, x, x1, nullptr, Mtok, Dm, Dm, 0);

    // 6) h2 = LN2(x1)
    ln_kernel<<<Mtok, 256, 0, stream>>>(x1, ln2_g, ln2_b, h2);

    // 7) u = gelu(h2 @ W_up^T + b_up)          (4096 x 4096 x 1024)
    gemm_bf16_kernel<<<dim3(4 * Dm / 256, Mtok / 128), 256, 0, stream>>>(
        h2, wU, b_up, nullptr, nullptr, ubuf, Mtok, 4 * Dm, Dm, 1);

    // 8) out = x1 + u @ W_down^T + b_down      (4096 x 1024 x 4096)
    gemm_bf16_kernel<<<dim3(Dm / 256, Mtok / 128), 256, 0, stream>>>(
        ubuf, wD, b_down, x1, out, nullptr, Mtok, Dm, 4 * Dm, 0);
}
